// RNN_55164559949996
// MI455X (gfx1250) — matrix operand order repaired and checked by reading, not compiled, not run on hardware
//
#include <hip/hip_runtime.h>
#include <hip/hip_bf16.h>

typedef __bf16 bf16_t;
typedef __attribute__((ext_vector_type(16))) __bf16 v16bf;
typedef __attribute__((ext_vector_type(8)))  __bf16 v8bf;
typedef __attribute__((ext_vector_type(8)))  float  v8f;
typedef __attribute__((ext_vector_type(4)))  unsigned int u32x4;
typedef __attribute__((ext_vector_type(8)))  unsigned int u32x8;

#define VOCABN 5000
#define HID    256
#define BATCH  32
#define SEQ    1024
#define NBLK_N 79        // decode N blocks of 64 cols
#define NPAD2  5056      // NBLK_N * 64
#define MTILES 2048      // (B*T)/16
#define MBLK_M 256       // decode M blocks of 128 rows (8 mtiles)

// ---------------------------------------------------------------------------
// Tensor Data Mover: issue a 2D row-major tile load (rows x 256 bf16) into LDS.
// D# built per CDNA5 ISA ch.8: group0 = {count,lds_addr,global_addr,type=2},
// group1 = {data_size=2B, tensor_dim0=256, tensor_dim1, tile_dim0=256,
// tile_dim1=rows, tensor_dim0_stride=256}.  2D tensor -> groups 2/3 omitted.
// Caller (one wave) must s_wait_tensorcnt(0) then barrier before LDS is read.
// ---------------------------------------------------------------------------
__device__ __forceinline__ void tdm_load_rows(const void* gsrc, void* ldst,
                                              unsigned rows, unsigned tensor_rows) {
  unsigned long long ga = (unsigned long long)(uintptr_t)gsrc;
  u32x4 g0;
  g0[0] = 1u;                                              // count=1, user mode
  g0[1] = (unsigned)(uintptr_t)ldst;                       // lds_addr
  g0[2] = (unsigned)ga;                                    // global_addr[31:0]
  g0[3] = ((unsigned)(ga >> 32) & 0x01FFFFFFu) | (2u << 30); // addr[56:32] | type=2
  u32x8 g1;
  g1[0] = 1u << 16;                                        // data_size = 2 bytes
  g1[1] = (unsigned)HID << 16;                             // tensor_dim0 = 256
  g1[2] = (tensor_rows & 0xFFFFu) << 16;                   // tensor_dim1 lo16
  g1[3] = ((tensor_rows >> 16) & 0xFFFFu) | ((unsigned)HID << 16); // dim1 hi | tile_dim0
  g1[4] = rows & 0xFFFFu;                                  // tile_dim1 (tile_dim2=0)
  g1[5] = (unsigned)HID;                                   // tensor_dim0_stride lo32
  g1[6] = 0u;                                              // stride hi | dim1_stride lo
  g1[7] = 0u;
  asm volatile("tensor_load_to_lds %0, %1" :: "s"(g0), "s"(g1) : "memory");
}

// ---------------------------------------------------------------------------
// Kernel 0: weight prep. W1 [H][2H] f32 -> Wx/Wh bf16 [N][K] (row [n][k] is
// already B-operand layout since y = z @ W1^T). Wd -> bf16 padded to NPAD2.
// ---------------------------------------------------------------------------
__global__ void prep_weights(const float* __restrict__ W1,
                             const float* __restrict__ Wd,
                             bf16_t* __restrict__ Wxb,
                             bf16_t* __restrict__ Whb,
                             bf16_t* __restrict__ Wdb) {
  int i = blockIdx.x * blockDim.x + threadIdx.x;
  const int total = HID * HID + NPAD2 * HID;
  for (; i < total; i += gridDim.x * blockDim.x) {
    if (i < HID * HID) {
      int n = i >> 8, k = i & 255;
      Wxb[i] = (bf16_t)W1[n * (2 * HID) + k];
      Whb[i] = (bf16_t)W1[n * (2 * HID) + HID + k];
    } else {
      int j = i - HID * HID;
      int n = j >> 8;
      Wdb[j] = (n < VOCABN) ? (bf16_t)Wd[j] : (bf16_t)0.0f;
    }
  }
}

// ---------------------------------------------------------------------------
// Kernel 1: embedding gather with padding_idx=0, f32 -> bf16.
// xeb row r = t*B + b (matches U-GEMM row ordering).
// ---------------------------------------------------------------------------
__global__ void embed_gather(const int* __restrict__ x,
                             const float* __restrict__ emb,
                             bf16_t* __restrict__ xeb) {
  long long i = (long long)blockIdx.x * blockDim.x + threadIdx.x;
  const long long total = (long long)SEQ * BATCH * HID;
  for (; i < total; i += (long long)gridDim.x * blockDim.x) {
    int k = (int)(i & 255);
    int r = (int)(i >> 8);          // r = t*32 + b
    int b = r & 31, t = r >> 5;
    int idx = x[b * SEQ + t];
    xeb[i] = (idx == 0) ? (bf16_t)0.0f : (bf16_t)emb[(long long)idx * HID + k];
  }
}

// ---------------------------------------------------------------------------
// Kernel 2: U = xeb @ Wx^T + b1   (M=32768, N=256, K=256), f32 output.
// ---------------------------------------------------------------------------
__global__ void __launch_bounds__(256) u_gemm(const bf16_t* __restrict__ A,
                                              const bf16_t* __restrict__ Bt,
                                              const float* __restrict__ bias,
                                              float* __restrict__ C) {
  const int lane = threadIdx.x & 31;
  const int wave = threadIdx.x >> 5;
  const int gw = blockIdx.x * 8 + wave;
  if (gw >= MTILES * 16) return;
  const int nt = gw & 15;
  const int mt = gw >> 4;
  const int half = lane >> 4;
  const int l15  = lane & 15;
  const long long arow = mt * 16 + l15;
  const long long brow = nt * 16 + l15;

  v8f c = {};
#pragma unroll
  for (int kt = 0; kt < HID; kt += 32) {
    const int ka = kt + 8 * half;
    v8bf lo = *(const v8bf*)(A + arow * HID + ka);
    v8bf hi = *(const v8bf*)(A + arow * HID + ka + 16);
    v16bf a = __builtin_shufflevector(lo, hi, 0,1,2,3,4,5,6,7,8,9,10,11,12,13,14,15);
    v16bf b = __builtin_shufflevector(*(const v8bf*)(Bt + brow * HID + ka), *(const v8bf*)(Bt + brow * HID + ka + 16), 0,1,2,3,4,5,6,7,8,9,10,11,12,13,14,15);
    c = __builtin_amdgcn_wmma_f32_16x16x32_bf16(false, a, false, b, (short)0, c,
                                                false, false);
  }
  const int n = nt * 16 + l15;
  const float bv = bias[n];
#pragma unroll
  for (int v = 0; v < 8; ++v) {
    int m = mt * 16 + v + 8 * half;
    C[(long long)m * HID + n] = c[v] + bv;
  }
}

// ---------------------------------------------------------------------------
// Kernel 3: persistent RNN scan on one WGP. 1024 threads = 32 waves covering
// the 2x16 tile grid of the per-step GEMM  h_new = tanh(U[t] + h @ Wh^T).
// Wh (128KB bf16, TDM-staged) + h double buffer (32KB) live in LDS.
// ---------------------------------------------------------------------------
__global__ void __launch_bounds__(1024) rnn_scan(const float* __restrict__ U,
                                                 const bf16_t* __restrict__ Whb,
                                                 bf16_t* __restrict__ hsb,
                                                 float* __restrict__ hT) {
  extern __shared__ char smem_raw[];
  bf16_t* whs  = (bf16_t*)smem_raw;             // [256][256]
  bf16_t* hbuf = whs + HID * HID;               // [2][32][256]

  const int tid = threadIdx.x;
  const int lane = tid & 31;
  const int wave = tid >> 5;

  // Stage Wh into LDS via Tensor Data Mover (one wave issues; TDM ignores EXEC)
  if (wave == 0) {
    tdm_load_rows(Whb, whs, HID, HID);
    __builtin_amdgcn_s_wait_tensorcnt(0);
  }
  // h0 = 0
  {
    v8bf z = {};
    for (int i = tid * 8; i < BATCH * HID; i += 1024 * 8)
      *(v8bf*)(hbuf + i) = z;
  }
  __syncthreads();

  const int mt = wave & 1;        // M tile (batch rows 0-15 / 16-31)
  const int nt = wave >> 1;       // N tile (hidden cols)
  const int half = lane >> 4;
  const int l15  = lane & 15;
  const int arow = mt * 16 + l15;
  const int brow = nt * 16 + l15;
  const int n = nt * 16 + l15;

  for (int t = 0; t < SEQ; ++t) {
    const bf16_t* hin = hbuf + (t & 1) * (BATCH * HID);
    bf16_t* hout      = hbuf + ((t + 1) & 1) * (BATCH * HID);

    // Prefetch next step's U tile rows while this step computes
    if (t + 1 < SEQ)
      __builtin_prefetch(U + ((long long)(t + 1) * BATCH + mt * 16 + 8 * half) * HID + n, 0, 0);

    // C <- U[t] tile (bias already folded in)
    v8f c;
#pragma unroll
    for (int v = 0; v < 8; ++v) {
      int m = mt * 16 + v + 8 * half;
      c[v] = U[((long long)t * BATCH + m) * HID + n];
    }
    // h @ Wh^T : 8 chained WMMAs, A from LDS h, B from LDS Wh
#pragma unroll
    for (int kt8 = 0; kt8 < 8; ++kt8) {
      const int ka = kt8 * 32 + 8 * half;
      v8bf lo = *(const v8bf*)(hin + arow * HID + ka);
      v8bf hi = *(const v8bf*)(hin + arow * HID + ka + 16);
      v16bf a = __builtin_shufflevector(lo, hi, 0,1,2,3,4,5,6,7,8,9,10,11,12,13,14,15);
      v16bf b = __builtin_shufflevector(*(const v8bf*)(whs + brow * HID + ka), *(const v8bf*)(whs + brow * HID + ka + 16), 0,1,2,3,4,5,6,7,8,9,10,11,12,13,14,15);
      c = __builtin_amdgcn_wmma_f32_16x16x32_bf16(false, a, false, b, (short)0, c,
                                                  false, false);
    }
    // tanh, publish h_new to LDS + hs to global
#pragma unroll
    for (int v = 0; v < 8; ++v) {
      float hv = tanhf(c[v]);
      int m = mt * 16 + v + 8 * half;       // batch row
      bf16_t hb = (bf16_t)hv;
      hout[m * HID + n] = hb;
      hsb[((long long)m * SEQ + t) * HID + n] = hb;   // row = b*T + t
      if (t == SEQ - 1) hT[m * HID + n] = hv;
    }
    __syncthreads();
  }
}

// ---------------------------------------------------------------------------
// Kernel 4: decode  out[b,t,v] = hs[b*T+t,:] . Wd[v,:] + bd[v]
// Block = 8 waves covers 128 rows x 64 cols. The 64x256 bf16 Wd panel is
// TDM-staged into LDS and shared by all waves; each wave computes 1 Mtile x
// 4 Ntiles, reusing its A fragment 4x. Nontemporal f32 stores (655MB stream).
// ---------------------------------------------------------------------------
__global__ void __launch_bounds__(256) decode_gemm(const bf16_t* __restrict__ A,
                                                   const bf16_t* __restrict__ Bt,
                                                   const float* __restrict__ bias,
                                                   float* __restrict__ out) {
  __shared__ bf16_t bstage[64 * HID];   // 32 KB Wd panel

  const int tid  = threadIdx.x;
  const int lane = tid & 31;
  const int wave = tid >> 5;                 // mtile within block
  const int mblk = blockIdx.x / NBLK_N;      // 0..255
  const int nblk = blockIdx.x % NBLK_N;      // 0..78

  if (wave == 0) {
    tdm_load_rows(Bt + (size_t)nblk * 64 * HID, bstage, 64, NPAD2);
    __builtin_amdgcn_s_wait_tensorcnt(0);
  }
  __syncthreads();

  const int half = lane >> 4;
  const int l15  = lane & 15;
  const long long arow = (long long)(mblk * 8 + wave) * 16 + l15;

  v8f acc[4] = {v8f{}, v8f{}, v8f{}, v8f{}};
#pragma unroll
  for (int kt = 0; kt < HID; kt += 32) {
    const int ka = kt + 8 * half;
    v8bf lo = *(const v8bf*)(A + arow * HID + ka);
    v8bf hi = *(const v8bf*)(A + arow * HID + ka + 16);
    v16bf a = __builtin_shufflevector(lo, hi, 0,1,2,3,4,5,6,7,8,9,10,11,12,13,14,15);
#pragma unroll
    for (int j = 0; j < 4; ++j) {
      v16bf b = __builtin_shufflevector(*(const v8bf*)(bstage + (j * 16 + l15) * HID + ka), *(const v8bf*)(bstage + (j * 16 + l15) * HID + ka + 16), 0,1,2,3,4,5,6,7,8,9,10,11,12,13,14,15);
      acc[j] = __builtin_amdgcn_wmma_f32_16x16x32_bf16(false, a, false, b, (short)0,
                                                       acc[j], false, false);
    }
  }

#pragma unroll
  for (int j = 0; j < 4; ++j) {
    const int n = nblk * 64 + j * 16 + l15;
    if (n < VOCABN) {
      const float bv = bias[n];
#pragma unroll
      for (int v = 0; v < 8; ++v) {
        long long m = (long long)(mblk * 8 + wave) * 16 + v + 8 * half;
        __builtin_nontemporal_store(acc[j][v] + bv, out + m * VOCABN + n);
      }
    }
  }
}

// ---------------------------------------------------------------------------
extern "C" void kernel_launch(void* const* d_in, const int* in_sizes, int n_in,
                              void* d_out, int out_size, void* d_ws, size_t ws_size,
                              hipStream_t stream) {
  const int*   x   = (const int*)d_in[0];
  const float* emb = (const float*)d_in[1];
  const float* W1  = (const float*)d_in[2];
  const float* b1  = (const float*)d_in[3];
  const float* Wd  = (const float*)d_in[4];
  const float* bd  = (const float*)d_in[5];

  float* out = (float*)d_out;
  float* hT  = out + (long long)BATCH * SEQ * VOCABN;

  // Workspace carve (~70 MB total)
  char* ws = (char*)d_ws;
  bf16_t* xeb = (bf16_t*)ws;  ws += (size_t)SEQ * BATCH * HID * sizeof(bf16_t);
  bf16_t* Wxb = (bf16_t*)ws;  ws += (size_t)HID * HID * sizeof(bf16_t);
  bf16_t* Whb = (bf16_t*)ws;  ws += (size_t)HID * HID * sizeof(bf16_t);
  bf16_t* Wdb = (bf16_t*)ws;  ws += (size_t)NPAD2 * HID * sizeof(bf16_t);
  bf16_t* hsb = (bf16_t*)ws;  ws += (size_t)SEQ * BATCH * HID * sizeof(bf16_t);
  float*  U   = (float*)ws;   // SEQ*BATCH*HID f32

  prep_weights<<<1024, 256, 0, stream>>>(W1, Wd, Wxb, Whb, Wdb);
  embed_gather<<<4096, 256, 0, stream>>>(x, emb, xeb);
  u_gemm<<<(MTILES * 16) / 8, 256, 0, stream>>>(xeb, Wxb, b1, U);
  rnn_scan<<<1, 1024, (HID * HID + 2 * BATCH * HID) * (int)sizeof(bf16_t), stream>>>(
      U, Whb, hsb, hT);
  decode_gemm<<<MBLK_M * NBLK_N, 256, 0, stream>>>(hsb, Wdb, bd, out);
}
